// Encoder_79542794322634
// MI455X (gfx1250) — matrix operand order repaired and checked by reading, not compiled, not run on hardware
//
#include <hip/hip_runtime.h>

typedef _Float16 v16h __attribute__((ext_vector_type(16)));
typedef _Float16 v8h  __attribute__((ext_vector_type(8)));
typedef _Float16 v4h  __attribute__((ext_vector_type(4)));
typedef float    v8f  __attribute__((ext_vector_type(8)));
typedef float    v4f  __attribute__((ext_vector_type(4)));

#define BB 64
#define TT 512
#define DD 1024
#define HH 1024
#define KTOT 2048   // D + H
#define NG 4096     // 4 gates * H

// workspace layout (bytes)
static const size_t OFF_X  = 0;                                   // BB*TT*DD f16   = 64 MB
static const size_t OFF_WT = OFF_X + (size_t)BB * TT * DD * 2;    // 2*NG*KTOT f16  = 32 MB
static const size_t OFF_H  = OFF_WT + (size_t)2 * NG * KTOT * 2;  // 2 dir * 2 parity * BB*HH f16
static const size_t OFF_C  = OFF_H + (size_t)2 * 2 * BB * HH * 2; // 2 dir * BB*HH f32

// ---------- prep: f32 -> f16 conversion of the input sequence ----------
__global__ void k_cvt_x(const float* __restrict__ x, _Float16* __restrict__ xf, int n4) {
  int i = blockIdx.x * blockDim.x + threadIdx.x;
  if (i < n4) {
    v4f v = ((const v4f*)x)[i];
    v4h o = { (_Float16)v.x, (_Float16)v.y, (_Float16)v.z, (_Float16)v.w };
    ((v4h*)xf)[i] = o;
  }
}

// ---------- prep: pack W [4][KTOT][HH] f32 -> WT[n'][k] f16, n' = hc*4 + g ----------
__global__ void k_pack_w(const float* __restrict__ W, _Float16* __restrict__ WT) {
  int tid = blockIdx.x * blockDim.x + threadIdx.x;  // NG*KTOT/8 threads
  int np  = tid >> 8;                               // KTOT/8 = 256 chunks per column
  int kb  = (tid & 255) << 3;
  int g = np & 3, hc = np >> 2;
  const float* src = W + (size_t)g * KTOT * HH + (size_t)kb * HH + hc;
  _Float16* dst = WT + (size_t)np * KTOT + kb;
#pragma unroll
  for (int j = 0; j < 8; ++j) dst[j] = (_Float16)src[(size_t)j * HH];
}

// ---------- prep: zero h-state (parity 0) and c-state ----------
__global__ void k_init(_Float16* __restrict__ h, float* __restrict__ c) {
  int i = blockIdx.x * blockDim.x + threadIdx.x;    // 2*BB*HH = 131072
  int dir = i >> 16, idx = i & 65535;
  h[(size_t)dir * (2 * BB * HH) + idx] = (_Float16)0.f;  // parity-0 buffer
  c[i] = 0.f;
}

// Async DMA: global -> LDS, 16 B per lane, tracked by ASYNCcnt.
// lds = raw 32-bit LDS byte address (low 32 bits of the generic pointer to a
// __shared__ object: addrspacecast AS3->flat is {aperture_hi, lds_offset}).
__device__ __forceinline__ void async_ld_b128(unsigned lds, const void* gaddr) {
  asm volatile("global_load_async_to_lds_b128 %0, %1, off"
               :: "v"(lds), "v"(gaddr) : "memory");
}

// ---------- one recurrent timestep: GEMM (WMMA f16->f32) + LSTM cell ----------
__global__ __launch_bounds__(128) void k_step(
    const _Float16* __restrict__ Xf, const _Float16* __restrict__ WT,
    _Float16* __restrict__ hst, float* __restrict__ cst,
    const float* __restrict__ bf, const float* __restrict__ bb,
    float* __restrict__ out, int t) {
  const int dir   = blockIdx.x >> 7;
  const int ntile = (blockIdx.x & 127) * 32;           // 32 packed columns = 8 h-cols * 4 gates
  const int tx    = dir ? (TT - 1 - t) : t;
  const float* bias = dir ? bb : bf;

  const _Float16* WTd   = WT + (size_t)dir * NG * KTOT;
  const _Float16* hread = hst + (size_t)dir * (2 * BB * HH) + (size_t)(t & 1) * (BB * HH);
  _Float16*       hwrit = hst + (size_t)dir * (2 * BB * HH) + (size_t)((t & 1) ^ 1) * (BB * HH);
  float*          cd    = cst + (size_t)dir * (BB * HH);

  const int wave = threadIdx.x >> 5;
  const int lane = threadIdx.x & 31;
  const int mrow = wave * 16 + (lane & 15);
  const int hi   = lane >> 4;              // half-wave select per ISA fragment layouts
  const int colq = (lane & 15) >> 2;       // 16B-chunk XOR swizzle key ((col>>2)&3)

  // A fragment (16x32 f16): lane = row; lanes 0-15 hold K {0..7,16..23}, 16-31 hold {8..15,24..31}
  const _Float16* aX = Xf + ((size_t)mrow * TT + tx) * DD + hi * 8;
  const _Float16* aH = hread + (size_t)mrow * HH + hi * 8;

  // B double buffer: per chunk [8 kkl][32 cols][64 B], 16B chunks XOR-swizzled by (col>>2)&3
  __shared__ __align__(16) char bbuf[2][16384];
  __shared__ float gt[64][33];

  v8f acc0 = {}, acc1 = {};
  union AU { v16h v; v8h h[2]; };

  // issue one chunk's async weight prefetch: 8 ops/wave, one 512B column slice each
  auto prefetch = [&](int chunk, int pb) {
    const size_t k0 = (size_t)chunk * 256;
#pragma unroll
    for (int i = 0; i < 8; ++i) {
      const int col = wave * 8 + i;  // 0..31 local column
      const _Float16* g = WTd + (size_t)(ntile + col) * KTOT + k0 + (size_t)lane * 8;
      unsigned laddr = (unsigned)(unsigned long long)(const void*)
          (&bbuf[pb][(lane >> 2) * 2048 + col * 64 + (((lane & 3) ^ ((col >> 2) & 3)) << 4)]);
      async_ld_b128(laddr, (const void*)g);
    }
  };

  prefetch(0, 0);
#pragma unroll 1
  for (int c = 0; c < 8; ++c) {
    prefetch((c + 1) & 7, (c + 1) & 1);              // next chunk into other buffer
    asm volatile("s_wait_asynccnt 0x8" ::: "memory"); // chunk c complete (only the 8 just issued remain)
    __syncthreads();                                  // all waves' slices visible

    const char* bB = bbuf[c & 1];
    const _Float16* arow = (c < 4) ? (aX + c * 256) : (aH + (c - 4) * 256);
    const _Float16* arown = (c < 3) ? (arow + 256) : ((c < 7) ? (aH + (c >= 4 ? (c - 3) : 0) * 256) : arow);
    __builtin_prefetch(arown, 0, 0);                  // global_prefetch_b8 for next A chunk

#pragma unroll
    for (int kkl = 0; kkl < 8; ++kkl) {
      AU a, w0, w1;
      a.h[0] = *(const v8h*)(arow + kkl * 32);
      a.h[1] = *(const v8h*)(arow + kkl * 32 + 16);
      const char* bk = bB + kkl * 2048;
      const int c0 = (lane & 15) * 64;
      const int c1 = (16 + (lane & 15)) * 64;
      w0.h[0] = *(const v8h*)(bk + c0 + (((hi + 0) ^ colq) << 4));
      w0.h[1] = *(const v8h*)(bk + c0 + (((hi + 2) ^ colq) << 4));
      w1.h[0] = *(const v8h*)(bk + c1 + (((hi + 0) ^ colq) << 4));
      w1.h[1] = *(const v8h*)(bk + c1 + (((hi + 2) ^ colq) << 4));
      acc0 = __builtin_amdgcn_wmma_f32_16x16x32_f16(false, a.v, false, w0.v, (short)0, acc0, false, false);
      acc1 = __builtin_amdgcn_wmma_f32_16x16x32_f16(false, a.v, false, w1.v, (short)0, acc1, false, false);
    }
    __syncthreads();                                  // all reads done before buffer reuse
  }

  // stage gate tile through LDS (C/D layout: VGPR r -> M = r + 8*hi; N = lane%16)
#pragma unroll
  for (int r = 0; r < 8; ++r) {
    gt[wave * 16 + hi * 8 + r][(lane & 15)]      = acc0[r];
    gt[wave * 16 + hi * 8 + r][16 + (lane & 15)] = acc1[r];
  }
  __syncthreads();

  const int hcb = ntile >> 2;                    // first h-column of this tile
  const size_t coff = (size_t)BB * TT * 2 * HH;  // c half of d_out
  for (int e = threadIdx.x; e < 512; e += 128) {
    int mm = e >> 3, hcl = e & 7;
    int hc = hcb + hcl;
    float gi = gt[mm][hcl * 4 + 0] + bias[hc];
    float gf = gt[mm][hcl * 4 + 1] + bias[HH + hc];
    float gc = gt[mm][hcl * 4 + 2] + bias[2 * HH + hc];
    float go = gt[mm][hcl * 4 + 3] + bias[3 * HH + hc];
    float ig = 1.f / (1.f + __expf(-gi));
    float fg = 1.f / (1.f + __expf(-gf));
    float cg = tanhf(gc);
    float og = 1.f / (1.f + __expf(-go));
    float cp = cd[mm * HH + hc];
    float cn = cp * fg + ig * cg;
    float hn = og * tanhf(cn);
    cd[mm * HH + hc] = cn;
    hwrit[mm * HH + hc] = (_Float16)hn;
    size_t ob = ((size_t)mm * TT + t) * (2 * HH) + (size_t)dir * HH + hc;
    out[ob] = hn;
    out[coff + ob] = cn;
  }
}

extern "C" void kernel_launch(void* const* d_in, const int* in_sizes, int n_in,
                              void* d_out, int out_size, void* d_ws, size_t ws_size,
                              hipStream_t stream) {
  (void)in_sizes; (void)n_in; (void)out_size; (void)ws_size;
  const float* x  = (const float*)d_in[0];
  const float* Wf = (const float*)d_in[1];
  const float* bf = (const float*)d_in[2];
  const float* Wb = (const float*)d_in[3];
  const float* bb = (const float*)d_in[4];
  float* out = (float*)d_out;

  char* ws = (char*)d_ws;
  _Float16* Xf  = (_Float16*)(ws + OFF_X);
  _Float16* WT  = (_Float16*)(ws + OFF_WT);
  _Float16* hst = (_Float16*)(ws + OFF_H);
  float*    cst = (float*)(ws + OFF_C);

  k_cvt_x<<<(BB * TT * DD / 4 + 255) / 256, 256, 0, stream>>>(x, Xf, BB * TT * DD / 4);
  k_pack_w<<<(NG * KTOT / 8) / 256, 256, 0, stream>>>(Wf, WT);
  k_pack_w<<<(NG * KTOT / 8) / 256, 256, 0, stream>>>(Wb, WT + (size_t)NG * KTOT);
  k_init<<<512, 256, 0, stream>>>(hst, cst);

  for (int t = 0; t < TT; ++t)
    k_step<<<256, 128, 0, stream>>>(Xf, WT, hst, cst, bf, bb, out, t);
}
